// CrossTableAttention_44341242364567
// MI455X (gfx1250) — matrix operand order repaired and checked by reading, not compiled, not run on hardware
//
#include <hip/hip_runtime.h>
#include <math.h>

// ---------------------------------------------------------------------------
// CrossTableAttention for MI455X (gfx1250, wave32)
//
// 1) Algebraic refactor: project each source table once (gather commutes with
//    the linear maps): 619 GFLOP -> ~138 GFLOP.
// 2) Precision: GEMMs are compute-bound (HBM floor ~6us/GEMM vs tens of us of
//    matrix math), so run them on V_WMMA_F32_16X16X32_F16 (8x the FLOP/issue
//    of the fp32 16x16x4 pipe) with fp32 accumulation. Inputs are O(1), so
//    f16 eps~5e-4 -> ~1e-3 relative error vs the fp32 reference.
// 3) Q/K/V stored f16 (L2-resident for the gathered attention reads);
//    softmax / biases / scaling in fp32; ctx overwrites Q in place.
// ---------------------------------------------------------------------------

typedef _Float16 v16h __attribute__((ext_vector_type(16)));
typedef _Float16 v8h  __attribute__((ext_vector_type(8)));
typedef _Float16 v4h  __attribute__((ext_vector_type(4)));
typedef _Float16 v2h  __attribute__((ext_vector_type(2)));
typedef float    v8f  __attribute__((ext_vector_type(8)));

#define T_TABLES 16
#define B_ROWS   1024
#define D_DIM    1024
#define R_REL    8
#define H_HEADS  16
#define DH_DIM   64
#define M_ROWS   (T_TABLES * B_ROWS)   // 16384

static __device__ __forceinline__ float wave_sum32(float v) {
#pragma unroll
    for (int off = 16; off > 0; off >>= 1)
        v += __shfl_xor(v, off, 32);
    return v;
}

static __device__ __forceinline__ v8f zero8() {
    v8f z = {0.f, 0.f, 0.f, 0.f, 0.f, 0.f, 0.f, 0.f};
    return z;
}

// Load a 16-half WMMA fragment from two 16-byte chunks at p+off0 / p+off1.
static __device__ __forceinline__ v16h frag16(const _Float16* p, int off0, int off1) {
    v16h f;
    ((v8h*)&f)[0] = *(const v8h*)(p + off0);
    ((v8h*)&f)[1] = *(const v8h*)(p + off1);
    return f;
}

// ---------------------------------------------------------------------------
// Kernel 0: fp32 -> f16 conversion, float4 vectorized (n % 4 == 0).
// ---------------------------------------------------------------------------
__global__ __launch_bounds__(256) void cvt_f16_kernel(
    const float* __restrict__ in, _Float16* __restrict__ out, int n4) {
    const int i = blockIdx.x * blockDim.x + threadIdx.x;
    if (i >= n4) return;
    const float4 v = ((const float4*)in)[i];
    v4h o = {(_Float16)v.x, (_Float16)v.y, (_Float16)v.z, (_Float16)v.w};
    ((v4h*)out)[i] = o;
}

// ---------------------------------------------------------------------------
// Kernel 1: rel_w[t,r] = sigmoid(rel_embs[t,r,:] . w_rel + b_rel)   (fp32)
// ---------------------------------------------------------------------------
__global__ __launch_bounds__(256) void relw_kernel(
    const float* __restrict__ rel_embs, const float* __restrict__ w_rel,
    const float* __restrict__ b_rel, float* __restrict__ rel_w) {
    const int t    = blockIdx.x;
    const int r    = threadIdx.x >> 5;
    const int lane = threadIdx.x & 31;
    const float* e = rel_embs + (t * R_REL + r) * D_DIM;
    float s = 0.f;
    for (int d = lane; d < D_DIM; d += 32) s += e[d] * w_rel[d];
    s = wave_sum32(s);
    if (lane == 0)
        rel_w[t * R_REL + r] = 1.f / (1.f + expf(-(s + b_rel[0])));
}

// ---------------------------------------------------------------------------
// Kernel 2: fused QKV projection, f16 WMMA.  Out[m,n] = sum_k X[m,k]*W[n,k].
// Wave: 64x16 output tile (4 M-tiles) for Q, K and V, sharing A fragments.
// A frag (16x32 f16): lane l -> row m+(l&15), K in [16h .. 16h+15]
// B frag (32x16 f16): lane l -> W row n+(l&15), K in [16h .. 16h+15]
// Q gets +bq (fp32) then f16; Kp/Vp bias-free (bias applied post-gather).
// ---------------------------------------------------------------------------
__global__ __launch_bounds__(256) void qkv_gemm_kernel(
    const _Float16* __restrict__ X,
    const _Float16* __restrict__ Wq, const float* __restrict__ bq,
    const _Float16* __restrict__ Wk, const _Float16* __restrict__ Wv,
    _Float16* __restrict__ Qh, _Float16* __restrict__ Kh,
    _Float16* __restrict__ Vh) {
    const int wave   = threadIdx.x >> 5;
    const int lane   = threadIdx.x & 31;
    const int lane15 = lane & 15;
    const int hi     = lane >> 4;
    const int n_base = blockIdx.x * 16;
    const int m_base = (blockIdx.y * 8 + wave) * 64;

    const _Float16* ap[4];
#pragma unroll
    for (int i = 0; i < 4; ++i)
        ap[i] = X + (m_base + 16 * i + lane15) * D_DIM + 16 * hi;
    const _Float16* bqp = Wq + (n_base + lane15) * D_DIM + 16 * hi;
    const _Float16* bkp = Wk + (n_base + lane15) * D_DIM + 16 * hi;
    const _Float16* bvp = Wv + (n_base + lane15) * D_DIM + 16 * hi;

    v8f aq[4], ak[4], av[4];
#pragma unroll
    for (int i = 0; i < 4; ++i) { aq[i] = zero8(); ak[i] = zero8(); av[i] = zero8(); }

#pragma unroll 2
    for (int k = 0; k < D_DIM; k += 32) {
        const v16h fbq = frag16(bqp, k, k + 8);
        const v16h fbk = frag16(bkp, k, k + 8);
        const v16h fbv = frag16(bvp, k, k + 8);
#pragma unroll
        for (int i = 0; i < 4; ++i) {
            const v16h fa = frag16(ap[i], k, k + 8);
            aq[i] = __builtin_amdgcn_wmma_f32_16x16x32_f16(
                false, fa, false, fbq, (short)0, aq[i], false, false);
            ak[i] = __builtin_amdgcn_wmma_f32_16x16x32_f16(
                false, fa, false, fbk, (short)0, ak[i], false, false);
            av[i] = __builtin_amdgcn_wmma_f32_16x16x32_f16(
                false, fa, false, fbv, (short)0, av[i], false, false);
        }
    }

    const int   c     = n_base + lane15;
    const float biasq = bq[c];
#pragma unroll
    for (int i = 0; i < 4; ++i) {
#pragma unroll
        for (int j = 0; j < 8; ++j) {
            const int row = m_base + 16 * i + j + (hi << 3);
            const int off = row * D_DIM + c;
            Qh[off] = (_Float16)(aq[i][j] + biasq);
            Kh[off] = (_Float16)ak[i][j];
            Vh[off] = (_Float16)av[i][j];
        }
    }
}

// ---------------------------------------------------------------------------
// Kernel 3: attention per (t,b). 16 waves = 16 heads, lane covers 2 dims.
// All math fp32; K/V gathered from f16 (L2-resident); ctx -> Qh in place.
// ---------------------------------------------------------------------------
__global__ __launch_bounds__(512) void attn_kernel(
    _Float16* __restrict__ Qh, const _Float16* __restrict__ Kh,
    const _Float16* __restrict__ Vh, const int* __restrict__ rel_idx,
    const float* __restrict__ relw, const float* __restrict__ bk,
    const float* __restrict__ bv) {
    const int tb   = blockIdx.x;
    const int t    = tb >> 10;          // B = 1024
    const int b    = tb & (B_ROWS - 1);
    const int h    = threadIdx.x >> 5;
    const int lane = threadIdx.x & 31;
    const int d0   = h * DH_DIM + 2 * lane;

    const int qoff = (t * B_ROWS + b) * D_DIM + d0;
    const v2h qh = *(const v2h*)(Qh + qoff);
    const float qx = (float)qh[0], qy = (float)qh[1];
    const float bkx = bk[d0], bky = bk[d0 + 1];
    const float bvx = bv[d0], bvy = bv[d0 + 1];

    int   idx[R_REL];
    float w[R_REL];
    float sc[R_REL];
#pragma unroll
    for (int r = 0; r < R_REL; ++r) {
        idx[r] = rel_idx[t * R_REL + r];
        w[r]   = relw[t * R_REL + r];
        const int ko = (idx[r] * B_ROWS + b) * D_DIM + d0;
        const v2h kh = *(const v2h*)(Kh + ko);
        const float kx = w[r] * (float)kh[0] + bkx;
        const float ky = w[r] * (float)kh[1] + bky;
        sc[r] = wave_sum32(qx * kx + qy * ky) * 0.125f;  // 1/sqrt(64)
    }

    float mx = sc[0];
#pragma unroll
    for (int r = 1; r < R_REL; ++r) mx = fmaxf(mx, sc[r]);
    float denom = 0.f;
#pragma unroll
    for (int r = 0; r < R_REL; ++r) { sc[r] = expf(sc[r] - mx); denom += sc[r]; }
    const float inv = 1.f / denom;

    float cx = 0.f, cy = 0.f;
#pragma unroll
    for (int r = 0; r < R_REL; ++r) {
        const int vo = (idx[r] * B_ROWS + b) * D_DIM + d0;
        const v2h vh = *(const v2h*)(Vh + vo);
        const float vx = w[r] * (float)vh[0] + bvx;
        const float vy = w[r] * (float)vh[1] + bvy;
        const float a  = sc[r] * inv;
        cx += a * vx;
        cy += a * vy;
    }
    v2h ch = {(_Float16)cx, (_Float16)cy};
    *(v2h*)(Qh + qoff) = ch;   // thread-private slot: safe in-place
}

// ---------------------------------------------------------------------------
// Kernel 4: out = ctx @ Wo^T + bo   (f16 WMMA, fp32 accumulate + store)
// ---------------------------------------------------------------------------
__global__ __launch_bounds__(256) void out_gemm_kernel(
    const _Float16* __restrict__ X, const _Float16* __restrict__ W,
    const float* __restrict__ bias, float* __restrict__ Out) {
    const int wave   = threadIdx.x >> 5;
    const int lane   = threadIdx.x & 31;
    const int lane15 = lane & 15;
    const int hi     = lane >> 4;
    const int n_base = blockIdx.x * 16;
    const int m_base = (blockIdx.y * 8 + wave) * 64;

    const _Float16* ap[4];
#pragma unroll
    for (int i = 0; i < 4; ++i)
        ap[i] = X + (m_base + 16 * i + lane15) * D_DIM + 16 * hi;
    const _Float16* bp = W + (n_base + lane15) * D_DIM + 16 * hi;

    v8f acc[4];
#pragma unroll
    for (int i = 0; i < 4; ++i) acc[i] = zero8();

#pragma unroll 2
    for (int k = 0; k < D_DIM; k += 32) {
        const v16h fb = frag16(bp, k, k + 8);
#pragma unroll
        for (int i = 0; i < 4; ++i) {
            const v16h fa = frag16(ap[i], k, k + 8);
            acc[i] = __builtin_amdgcn_wmma_f32_16x16x32_f16(
                false, fa, false, fb, (short)0, acc[i], false, false);
        }
    }

    const int   c  = n_base + lane15;
    const float bb = bias[c];
#pragma unroll
    for (int i = 0; i < 4; ++i) {
#pragma unroll
        for (int j = 0; j < 8; ++j) {
            const int row = m_base + 16 * i + j + (hi << 3);
            Out[row * D_DIM + c] = acc[i][j] + bb;
        }
    }
}

// ---------------------------------------------------------------------------
extern "C" void kernel_launch(void* const* d_in, const int* in_sizes, int n_in,
                              void* d_out, int out_size, void* d_ws, size_t ws_size,
                              hipStream_t stream) {
    const float* table_embs = (const float*)d_in[0];
    const float* rel_embs   = (const float*)d_in[1];
    const int*   rel_idx    = (const int*)d_in[2];
    const float* Wq = (const float*)d_in[3];
    const float* bq = (const float*)d_in[4];
    const float* Wk = (const float*)d_in[5];
    const float* bk = (const float*)d_in[6];
    const float* Wv = (const float*)d_in[7];
    const float* bv = (const float*)d_in[8];
    const float* Wo = (const float*)d_in[9];
    const float* bo = (const float*)d_in[10];
    const float* w_rel = (const float*)d_in[11];
    const float* b_rel = (const float*)d_in[12];
    float* out = (float*)d_out;

    char* ws = (char*)d_ws;
    const size_t hMat = (size_t)M_ROWS * D_DIM * sizeof(_Float16);  // 32 MB
    const size_t hW   = (size_t)D_DIM * D_DIM * sizeof(_Float16);   //  2 MB
    _Float16* Xh  = (_Float16*)(ws);
    _Float16* Qh  = (_Float16*)(ws + 1 * hMat);   // Q, then ctx in-place
    _Float16* Kh  = (_Float16*)(ws + 2 * hMat);
    _Float16* Vh  = (_Float16*)(ws + 3 * hMat);
    _Float16* Wqh = (_Float16*)(ws + 4 * hMat);
    _Float16* Wkh = (_Float16*)(ws + 4 * hMat + 1 * hW);
    _Float16* Wvh = (_Float16*)(ws + 4 * hMat + 2 * hW);
    _Float16* Woh = (_Float16*)(ws + 4 * hMat + 3 * hW);
    float*    relw = (float*)(ws + 4 * hMat + 4 * hW);

    const int nX4 = M_ROWS * D_DIM / 4;   // 4.19M float4's
    const int nW4 = D_DIM * D_DIM / 4;    // 262144
    cvt_f16_kernel<<<nX4 / 256, 256, 0, stream>>>(table_embs, Xh, nX4);
    cvt_f16_kernel<<<nW4 / 256, 256, 0, stream>>>(Wq, Wqh, nW4);
    cvt_f16_kernel<<<nW4 / 256, 256, 0, stream>>>(Wk, Wkh, nW4);
    cvt_f16_kernel<<<nW4 / 256, 256, 0, stream>>>(Wv, Wvh, nW4);
    cvt_f16_kernel<<<nW4 / 256, 256, 0, stream>>>(Wo, Woh, nW4);

    relw_kernel<<<T_TABLES, 256, 0, stream>>>(rel_embs, w_rel, b_rel, relw);
    qkv_gemm_kernel<<<dim3(D_DIM / 16, M_ROWS / 512), 256, 0, stream>>>(
        Xh, Wqh, bq, Wkh, Wvh, Qh, Kh, Vh);
    attn_kernel<<<T_TABLES * B_ROWS, 512, 0, stream>>>(
        Qh, Kh, Vh, rel_idx, relw, bk, bv);
    out_gemm_kernel<<<dim3(D_DIM / 16, M_ROWS / 512), 256, 0, stream>>>(
        Qh, Woh, bo, out);
}
